// Transformer_Proposed_2_58437325029417
// MI455X (gfx1250) — matrix operand order repaired and checked by reading, not compiled, not run on hardware
//
#include <hip/hip_runtime.h>
#include <hip/hip_bf16.h>

typedef __attribute__((ext_vector_type(16))) _Float16 v16h;
typedef __attribute__((ext_vector_type(8)))  _Float16 v8h;
typedef __attribute__((ext_vector_type(8)))  float    v8f;
typedef __attribute__((ext_vector_type(4)))  float    v4f;

#define NBATCH 512
#define NFRM   81
#define NPTS   17
#define NCIN   2
#define NK27   27
#define DIMS   32
#define DIMT   544
#define HIDT   1088
#define SEQMX  54

__device__ __forceinline__ float gelu_exact(float v) {
  return 0.5f * v * (1.0f + erff(v * 0.70710678118654752f));
}

__device__ __forceinline__ v16h mkfrag(v8h lo, v8h hi) {
  return __builtin_shufflevector(lo, hi, 0, 1, 2, 3, 4, 5, 6, 7, 8, 9, 10, 11, 12, 13, 14, 15);
}

// ---------------- weight conversion with transpose: Wt[n*K+k] = W[k*N+n] ----------------
__global__ void k_f32_to_f16_t(const float* __restrict__ s, _Float16* __restrict__ d, int K, int N) {
  int i = blockIdx.x * blockDim.x + threadIdx.x;
  if (i >= K * N) return;
  int n = i / K, k = i % K;
  d[i] = (_Float16)s[(size_t)k * N + n];
}

// ---------------- DCT matrices ----------------
__global__ void k_init_dct(float* __restrict__ M2f, float* __restrict__ M2, float* __restrict__ M3) {
  int i = blockIdx.x * blockDim.x + threadIdx.x;
  const float PI = 3.14159265358979323846f;
  if (i < 27 * 81) {
    int k = i / 81, f = i % 81;
    M2f[i] = 2.0f * cosf(PI * (float)(k * (2 * f + 1)) / 162.0f);
  }
  if (i < 27 * 27) {
    int k = i / 27, m = i % 27;
    M2[i] = 2.0f * cosf(PI * (float)(k * (2 * m + 1)) / 54.0f);
    M3[i] = (m == 0) ? 1.0f : 2.0f * cosf(PI * (float)(m * (2 * k + 1)) / 54.0f);
  }
}

// ---------------- WMMA GEMM: D = act(A@W + bias) (+resid) ----------------
// A: MxK f32 row-major.  Wt: NxK f16 (pre-transposed).  D: MxN f32.
// Block tile 128 x BN, 8 waves (4 row x 2 col), wave tile 32 x (BN/2).
// B tile is copied global->LDS with GLOBAL_LOAD_ASYNC_TO_LDS_B128 (ASYNCcnt),
// overlapping the VGPR-path f32->f16 conversion of the A tile.
template <int BN>
__global__ __launch_bounds__(256) void k_gemm_wmma(
    const float* __restrict__ A, const _Float16* __restrict__ Wt,
    const float* __restrict__ bias, const float* __restrict__ resid,
    float* __restrict__ D, int M, int N, int K, int act)
{
  constexpr int BM = 128, BK = 32;
  constexpr int WN = BN / 2;
  constexpr int NJ = WN / 16;
  constexpr int LDR = BK + 8;  // 40 halves = 80B row stride (16B aligned)
  __shared__ _Float16 sA[BM][LDR];
  __shared__ _Float16 sB[BN][LDR];  // sB[n][k]
  const int tid  = threadIdx.x;
  const int wave = tid >> 5, lane = tid & 31;
  const int wr = wave >> 1, wc = wave & 1;
  const int row = lane & 15, half = lane >> 4;
  const int m0 = blockIdx.y * BM, n0 = blockIdx.x * BN;
  const bool fullM = (m0 + BM <= M);
  const bool fullN = (n0 + BN <= N);
  const bool kA = ((K & 7) == 0);
  v8f acc[2][NJ];
#pragma unroll
  for (int a = 0; a < 2; ++a)
#pragma unroll
    for (int j = 0; j < NJ; ++j) acc[a][j] = {};

  for (int k0 = 0; k0 < K; k0 += BK) {
    const bool kfull = kA && (k0 + BK <= K);
    // ---- B tile first: BN x BK from Wt (N-major): async DMA straight to LDS ----
#pragma unroll
    for (int cc = 0; cc < BN / 64; ++cc) {
      int chunk = tid + cc * 256;
      int c = chunk >> 2, ck = (chunk & 3) * 8;
      if (kfull && fullN) {
        const _Float16* gp = Wt + (size_t)(n0 + c) * K + (k0 + ck);
        unsigned lds = (unsigned)(size_t)&sB[c][ck];
        unsigned long long ga = (unsigned long long)(size_t)gp;
        asm volatile("global_load_async_to_lds_b128 %0, %1, off"
                     :: "v"(lds), "v"(ga) : "memory");
      } else {
        v8h hv;
#pragma unroll
        for (int i = 0; i < 8; ++i) {
          int gn = n0 + c, gk = k0 + ck + i;
          hv[i] = (gn < N && gk < K) ? Wt[(size_t)gn * K + gk] : (_Float16)0.0f;
        }
        *(v8h*)&sB[c][ck] = hv;
      }
    }
    // ---- A tile: BM x BK (f32 -> f16 through VGPRs), 512 chunks of 8, 2 per thread ----
#pragma unroll
    for (int cc = 0; cc < 2; ++cc) {
      int chunk = tid * 2 + cc;
      int r = chunk >> 2, c0 = (chunk & 3) * 8;
      const float* ap = A + (size_t)(m0 + r) * K + (k0 + c0);
      v8h hv;
      if (kfull && fullM) {
        v4f x0 = *(const v4f*)ap;
        v4f x1 = *(const v4f*)(ap + 4);
#pragma unroll
        for (int i = 0; i < 4; ++i) { hv[i] = (_Float16)x0[i]; hv[4 + i] = (_Float16)x1[i]; }
        if (k0 + BK < K) __builtin_prefetch(ap + BK, 0, 0);
      } else {
#pragma unroll
        for (int i = 0; i < 8; ++i) {
          int gm = m0 + r, gk = k0 + c0 + i;
          hv[i] = (gm < M && gk < K) ? (_Float16)A[(size_t)gm * K + gk] : (_Float16)0.0f;
        }
      }
      *(v8h*)&sA[r][c0] = hv;
    }
    asm volatile("s_wait_asynccnt 0x0" ::: "memory");
    __syncthreads();
    // ---- fragments: all contiguous 16B LDS reads ----
    v16h afr[2], bfr[NJ];
#pragma unroll
    for (int a = 0; a < 2; ++a) {
      const _Float16* p = &sA[wr * 32 + a * 16 + row][0];
      v8h lo = *(const v8h*)(p + half * 8);
      v8h hi = *(const v8h*)(p + 16 + half * 8);
      afr[a] = mkfrag(lo, hi);
    }
#pragma unroll
    for (int j = 0; j < NJ; ++j) {
      const _Float16* p = &sB[wc * WN + j * 16 + row][0];
      v8h lo = *(const v8h*)(p + half * 8);
      v8h hi = *(const v8h*)(p + 16 + half * 8);
      bfr[j] = mkfrag(lo, hi);
    }
#pragma unroll
    for (int a = 0; a < 2; ++a)
#pragma unroll
      for (int j = 0; j < NJ; ++j)
        acc[a][j] = __builtin_amdgcn_wmma_f32_16x16x32_f16(
            false, afr[a], false, bfr[j], (short)0, acc[a][j], false, false);
    __syncthreads();
  }
  // ---- epilogue ----
#pragma unroll
  for (int a = 0; a < 2; ++a) {
#pragma unroll
    for (int r = 0; r < 8; ++r) {
      int gm = m0 + wr * 32 + a * 16 + half * 8 + r;
      if (gm >= M) continue;
      size_t base = (size_t)gm * N;
#pragma unroll
      for (int j = 0; j < NJ; ++j) {
        int gn = n0 + wc * WN + j * 16 + row;
        if (gn < N) {
          float v = acc[a][j][r] + (bias ? bias[gn] : 0.0f);
          if (act) v = gelu_exact(v);
          if (resid) v += resid[base + gn];
          D[base + gn] = v;
        }
      }
    }
  }
}

// ---------------- LayerNorm (one row per block) ----------------
__global__ __launch_bounds__(256) void k_ln(
    const float* __restrict__ src, float* __restrict__ dst,
    const float* __restrict__ g, const float* __restrict__ b, int C, float eps)
{
  __shared__ float red[256];
  const int t = threadIdx.x, BD = blockDim.x;
  const size_t base = (size_t)blockIdx.x * C;
  const float* xr = src + base;
  float s = 0.f;
  for (int i = t; i < C; i += BD) s += xr[i];
  red[t] = s; __syncthreads();
  for (int k = BD >> 1; k > 0; k >>= 1) { if (t < k) red[t] += red[t + k]; __syncthreads(); }
  const float mu = red[0] / (float)C;
  __syncthreads();
  float v = 0.f;
  for (int i = t; i < C; i += BD) { float d = xr[i] - mu; v += d * d; }
  red[t] = v; __syncthreads();
  for (int k = BD >> 1; k > 0; k >>= 1) { if (t < k) red[t] += red[t + k]; __syncthreads(); }
  const float inv = rsqrtf(red[0] / (float)C + eps);
  float* yr = dst + base;
  for (int i = t; i < C; i += BD) yr[i] = (xr[i] - mu) * inv * g[i] + b[i];
}

// ---------------- spatial embed ----------------
__global__ void k_embed_sp(const float* __restrict__ x, const float* __restrict__ jw,
                           const float* __restrict__ jb, const float* __restrict__ pos,
                           float* __restrict__ xs)
{
  int i = blockIdx.x * blockDim.x + threadIdx.x;
  const int total = NBATCH * NK27 * NPTS * DIMS;
  if (i >= total) return;
  int o = i & 31;
  int p = (i >> 5) % NPTS;
  int s = i / (NPTS * DIMS);
  int b = s / NK27, f = (s % NK27) + 27;
  const float* xb = x + (((size_t)b * NFRM + f) * NPTS + p) * NCIN;
  xs[i] = xb[0] * jw[o] + xb[1] * jw[32 + o] + jb[o] + pos[p * 32 + o];
}

// ---------------- spatial attention: dim32, 8 heads, hd=4, seq 17 ----------------
__global__ __launch_bounds__(32) void k_attn_sp(const float* __restrict__ qkv, float* __restrict__ out)
{
  const int s = blockIdx.x >> 3;
  const int h = blockIdx.x & 7;
  __shared__ float q[17][4], kk[17][4], vv[17][4];
  const int t = threadIdx.x;
  for (int i = t; i < 17 * 4; i += 32) {
    int p = i >> 2, d = i & 3;
    const float* base = qkv + ((size_t)s * 17 + p) * 96;
    q[p][d]  = base[h * 4 + d];
    kk[p][d] = base[32 + h * 4 + d];
    vv[p][d] = base[64 + h * 4 + d];
  }
  __syncthreads();
  if (t < 17) {
    float sc[17];
    float mx = -1e30f;
    for (int j = 0; j < 17; ++j) {
      float sum = 0.f;
      for (int d = 0; d < 4; ++d) sum += q[t][d] * kk[j][d];
      sum *= 0.5f;  // 4^-0.5
      sc[j] = sum; mx = fmaxf(mx, sum);
    }
    float den = 0.f;
    for (int j = 0; j < 17; ++j) { sc[j] = expf(sc[j] - mx); den += sc[j]; }
    float inv = 1.f / den;
    float o0 = 0, o1 = 0, o2 = 0, o3 = 0;
    for (int j = 0; j < 17; ++j) {
      float w = sc[j] * inv;
      o0 += w * vv[j][0]; o1 += w * vv[j][1]; o2 += w * vv[j][2]; o3 += w * vv[j][3];
    }
    float* ob = out + ((size_t)s * 17 + t) * 32 + h * 4;
    ob[0] = o0; ob[1] = o1; ob[2] = o2; ob[3] = o3;
  }
}

// ---------------- mix attention: dim544, 8 heads, hd=68, seq 54 ----------------
__global__ __launch_bounds__(64) void k_attn_mix(const float* __restrict__ qkv, float* __restrict__ out)
{
  const int b = blockIdx.x >> 3;
  const int h = blockIdx.x & 7;
  __shared__ float sk[54][68];
  __shared__ float sv[54][68];
  __shared__ float sq[68];
  __shared__ float sp[54];
  __shared__ float red[64];
  const int t = threadIdx.x;
  const size_t rowbase = (size_t)b * SEQMX;
  for (int i = t; i < 54 * 68; i += 64) {
    int j = i / 68, d = i % 68;
    const float* base = qkv + (rowbase + j) * 1632;
    sk[j][d] = base[544  + h * 68 + d];
    sv[j][d] = base[1088 + h * 68 + d];
  }
  __syncthreads();
  const float scale = rsqrtf(68.0f);
  for (int qi = 0; qi < 54; ++qi) {
    const float* qb = qkv + (rowbase + qi) * 1632 + h * 68;
    for (int i = t; i < 68; i += 64) sq[i] = qb[i];
    __syncthreads();
    float sc = -1e30f;
    if (t < 54) {
      float sum = 0.f;
      for (int d = 0; d < 68; ++d) sum += sq[d] * sk[t][d];
      sc = sum * scale;
    }
    red[t] = sc; __syncthreads();
    for (int s = 32; s > 0; s >>= 1) { if (t < s) red[t] = fmaxf(red[t], red[t + s]); __syncthreads(); }
    const float mx = red[0]; __syncthreads();
    float e = (t < 54) ? expf(sc - mx) : 0.f;
    if (t < 54) sp[t] = e;
    red[t] = e; __syncthreads();
    for (int s = 32; s > 0; s >>= 1) { if (t < s) red[t] += red[t + s]; __syncthreads(); }
    const float inv = 1.0f / red[0]; __syncthreads();
    float* ob = out + (rowbase + qi) * DIMT + h * 68;
    for (int d = t; d < 68; d += 64) {
      float acc = 0.f;
      for (int j = 0; j < 54; ++j) acc += sp[j] * sv[j][d];
      ob[d] = acc * inv;
    }
    __syncthreads();
  }
}

// ---------------- DCT over 81 frames ----------------
__global__ void k_dct_frames(const float* __restrict__ x, const float* __restrict__ M2f,
                             float* __restrict__ xf)
{
  int i = blockIdx.x * blockDim.x + threadIdx.x;
  const int total = NBATCH * NK27 * NPTS * NCIN;
  if (i >= total) return;
  int pc = i % 34;
  int k  = (i / 34) % NK27;
  int b  = i / (34 * NK27);
  const float* xb = x + (size_t)b * NFRM * 34 + pc;
  const float* m = M2f + k * NFRM;
  float s = 0.f;
  for (int f = 0; f < NFRM; ++f) s += xb[(size_t)f * 34] * m[f];
  xf[i] = s;
}

// ---------------- seq-27 matrix contract ----------------
__global__ void k_seq_mat27(const float* __restrict__ in, const float* __restrict__ Mm,
                            float* __restrict__ out)
{
  int i = blockIdx.x * blockDim.x + threadIdx.x;
  const int total = NBATCH * NK27 * DIMT;
  if (i >= total) return;
  int c = i % DIMT;
  int k = (i / DIMT) % NK27;
  int b = i / (DIMT * NK27);
  const float* xb = in + ((size_t)b * NK27) * DIMT + c;
  const float* m = Mm + k * NK27;
  float s = 0.f;
#pragma unroll
  for (int f = 0; f < NK27; ++f) s += xb[(size_t)f * DIMT] * m[f];
  out[i] = s;
}

// ---------------- concat xt ----------------
__global__ void k_concat_xt(const float* __restrict__ xf_emb, const float* __restrict__ sf,
                            const float* __restrict__ tpos_, const float* __restrict__ tpos,
                            float* __restrict__ xt)
{
  int i = blockIdx.x * blockDim.x + threadIdx.x;
  const int total = NBATCH * SEQMX * DIMT;
  if (i >= total) return;
  int c = i % DIMT;
  int r = i / DIMT;
  int b = r / SEQMX, f = r % SEQMX;
  float v;
  if (f < NK27) v = xf_emb[((size_t)b * NK27 + f) * DIMT + c] + tpos_[f * DIMT + c];
  else { int gidx = f - NK27; v = sf[((size_t)b * NK27 + gidx) * DIMT + c] + tpos[gidx * DIMT + c]; }
  xt[i] = v;
}

// ---------------- half copy / scatter-add ----------------
__global__ void k_copy_half(const float* __restrict__ xt, float* __restrict__ dst, int off)
{
  int i = blockIdx.x * blockDim.x + threadIdx.x;
  const int total = NBATCH * NK27 * DIMT;
  if (i >= total) return;
  int c = i % DIMT; int r = i / DIMT;
  int b = r / NK27, f = r % NK27;
  dst[i] = xt[((size_t)b * SEQMX + off + f) * DIMT + c];
}
__global__ void k_add_half(float* __restrict__ xt, const float* __restrict__ src, int off)
{
  int i = blockIdx.x * blockDim.x + threadIdx.x;
  const int total = NBATCH * NK27 * DIMT;
  if (i >= total) return;
  int c = i % DIMT; int r = i / DIMT;
  int b = r / NK27, f = r % NK27;
  xt[((size_t)b * SEQMX + off + f) * DIMT + c] += src[i];
}

// ---------------- patches ----------------
__global__ void k_patches(const float* __restrict__ xtn, const float* __restrict__ pw,
                          const float* __restrict__ pb, float* __restrict__ cat)
{
  int i = blockIdx.x * blockDim.x + threadIdx.x;
  const int total = NBATCH * NFRM * 2 * DIMT;
  if (i >= total) return;
  int c2 = i % (2 * DIMT);
  int o  = (i / (2 * DIMT)) % NFRM;
  int b  = i / (2 * DIMT * NFRM);
  int half = (c2 >= DIMT) ? 1 : 0;
  int c = c2 - half * DIMT;
  const float* xb = xtn + ((size_t)b * SEQMX + half * NK27) * DIMT + c;
  const float* m = pw + o * NK27;
  float s = 0.f;
#pragma unroll
  for (int j = 0; j < NK27; ++j) s += xb[(size_t)j * DIMT] * m[j];
  cat[i] = s + pb[o];
}

// =====================================================================
extern "C" void kernel_launch(void* const* d_in, const int* in_sizes, int n_in,
                              void* d_out, int out_size, void* d_ws, size_t ws_size,
                              hipStream_t stream) {
  (void)n_in; (void)out_size; (void)ws_size;
  const float** Fp = (const float**)d_in;

  struct SpB { const float *ln1_g,*ln1_b,*qkv_w,*qkv_b,*proj_w,*proj_b,*ln2_g,*ln2_b,*fc1_w,*fc1_b,*fc2_w,*fc2_b; };
  struct MxB { const float *ln1_g,*ln1_b,*qkv_w,*qkv_b,*proj_w,*proj_b,*ln2_g,*ln2_b,*fc1_w,*fc1_b,*fc2_w,*fc2_b,
                           *ln3_g,*ln3_b,*f1_w,*f1_b,*f2_w,*f2_b; };
  const float *x,*joint_w,*joint_b,*freq_w,*freq_b,*spatial_pos,*temp_pos,*temp_pos_;
  const float *sp_norm_g,*sp_norm_b,*t_norm_g,*t_norm_b,*patches_w,*patches_b;
  const float *head_ln_g,*head_ln_b,*head_w,*head_b;
  SpB sp[4]; MxB mx[4];

  const int XSZ = NBATCH * NFRM * NPTS * NCIN;
  if (in_sizes[0] == XSZ) {
    // dict insertion order, x first
    int q = 0;
    x = Fp[q++];
    joint_w = Fp[q++]; joint_b = Fp[q++]; freq_w = Fp[q++]; freq_b = Fp[q++];
    spatial_pos = Fp[q++]; temp_pos = Fp[q++]; temp_pos_ = Fp[q++];
    for (int i = 0; i < 4; ++i) { SpB& s = sp[i];
      s.ln1_g=Fp[q++]; s.ln1_b=Fp[q++]; s.qkv_w=Fp[q++]; s.qkv_b=Fp[q++]; s.proj_w=Fp[q++]; s.proj_b=Fp[q++];
      s.ln2_g=Fp[q++]; s.ln2_b=Fp[q++]; s.fc1_w=Fp[q++]; s.fc1_b=Fp[q++]; s.fc2_w=Fp[q++]; s.fc2_b=Fp[q++]; }
    for (int i = 0; i < 4; ++i) { MxB& m = mx[i];
      m.ln1_g=Fp[q++]; m.ln1_b=Fp[q++]; m.qkv_w=Fp[q++]; m.qkv_b=Fp[q++]; m.proj_w=Fp[q++]; m.proj_b=Fp[q++];
      m.ln2_g=Fp[q++]; m.ln2_b=Fp[q++]; m.fc1_w=Fp[q++]; m.fc1_b=Fp[q++]; m.fc2_w=Fp[q++]; m.fc2_b=Fp[q++];
      m.ln3_g=Fp[q++]; m.ln3_b=Fp[q++]; m.f1_w=Fp[q++]; m.f1_b=Fp[q++]; m.f2_w=Fp[q++]; m.f2_b=Fp[q++]; }
    sp_norm_g=Fp[q++]; sp_norm_b=Fp[q++]; t_norm_g=Fp[q++]; t_norm_b=Fp[q++];
    patches_w=Fp[q++]; patches_b=Fp[q++]; head_ln_g=Fp[q++]; head_ln_b=Fp[q++]; head_w=Fp[q++]; head_b=Fp[q++];
  } else {
    // jax tree_leaves sorted-key order, x last
    int q = 0;
    freq_b=Fp[q++]; freq_w=Fp[q++]; head_b=Fp[q++]; head_ln_b=Fp[q++]; head_ln_g=Fp[q++]; head_w=Fp[q++];
    joint_b=Fp[q++]; joint_w=Fp[q++];
    for (int i = 0; i < 4; ++i) { MxB& m = mx[i];
      m.f1_b=Fp[q++]; m.f1_w=Fp[q++]; m.f2_b=Fp[q++]; m.f2_w=Fp[q++]; m.fc1_b=Fp[q++]; m.fc1_w=Fp[q++];
      m.fc2_b=Fp[q++]; m.fc2_w=Fp[q++]; m.ln1_b=Fp[q++]; m.ln1_g=Fp[q++]; m.ln2_b=Fp[q++]; m.ln2_g=Fp[q++];
      m.ln3_b=Fp[q++]; m.ln3_g=Fp[q++]; m.proj_b=Fp[q++]; m.proj_w=Fp[q++]; m.qkv_b=Fp[q++]; m.qkv_w=Fp[q++]; }
    patches_b=Fp[q++]; patches_w=Fp[q++];
    for (int i = 0; i < 4; ++i) { SpB& s = sp[i];
      s.fc1_b=Fp[q++]; s.fc1_w=Fp[q++]; s.fc2_b=Fp[q++]; s.fc2_w=Fp[q++]; s.ln1_b=Fp[q++]; s.ln1_g=Fp[q++];
      s.ln2_b=Fp[q++]; s.ln2_g=Fp[q++]; s.proj_b=Fp[q++]; s.proj_w=Fp[q++]; s.qkv_b=Fp[q++]; s.qkv_w=Fp[q++]; }
    sp_norm_b=Fp[q++]; sp_norm_g=Fp[q++]; spatial_pos=Fp[q++]; t_norm_b=Fp[q++]; t_norm_g=Fp[q++];
    temp_pos=Fp[q++]; temp_pos_=Fp[q++]; x=Fp[q++];
  }

  // ---------------- workspace layout ----------------
  char* wsb = (char*)d_ws;
  _Float16* W16 = (_Float16*)wsb;                 // 32MB f16 weight pool (N-major)
  const size_t OFF_DCT = 32ull << 20;
  float* dM2f = (float*)(wsb + OFF_DCT);          // 27*81
  float* dM2  = dM2f + 27 * 81;
  float* dM3  = dM2  + 27 * 27;
  const size_t P0 = OFF_DCT + (64ull << 10);
  float* buf0 = (float*)(wsb + P0);               // xs        (235008*32)
  float* buf1 = buf0 + 7520256;                   // ln/attn scratch (spatial), sf_norm
  float* buf2 = buf1 + 7520256;                   // qkv spatial (235008*96) / mix ln+attn (27648*544)
  float* buf3 = buf2 + 22560768;                  // spatial hid / mix half copies (2 x 13824*544)
  float* buf4 = buf3 + 15040512;                  // xt (27648*544)
  float* buf5 = buf4 + 15040512;                  // qkv mix (27648*1632) / cat (41472*1088)
  float* buf6 = buf5 + 45121536;                  // mix hidden (13824*1088)
  float* buf7 = buf6 + 15040512;                  // freq-branch temporaries (2 x 13824*544)
  float* buf3b = buf3 + 7520256;
  float* buf7b = buf7 + 7520256;

  // ---------------- convert weights to f16 (transposed to N-major) ----------------
  size_t woff = 0;
  auto cvt = [&](const float* srcp, int K, int N) -> size_t {
    size_t off = woff;
    size_t n = (size_t)K * N;
    woff += (n + 127) & ~(size_t)127;
    k_f32_to_f16_t<<<(int)((n + 255) / 256), 256, 0, stream>>>(srcp, W16 + off, K, N);
    return off;
  };
  size_t w_freq = cvt(freq_w, 34, DIMT);
  size_t w_head = cvt(head_w, 1088, 51);
  size_t wsp_qkv[4], wsp_proj[4], wsp_fc1[4], wsp_fc2[4];
  for (int i = 0; i < 4; ++i) {
    wsp_qkv[i]  = cvt(sp[i].qkv_w, 32, 96);
    wsp_proj[i] = cvt(sp[i].proj_w, 32, 32);
    wsp_fc1[i]  = cvt(sp[i].fc1_w, 32, 64);
    wsp_fc2[i]  = cvt(sp[i].fc2_w, 64, 32);
  }
  size_t wmx_qkv[4], wmx_proj[4], wmx_fc1[4], wmx_fc2[4], wmx_f1[4], wmx_f2[4];
  for (int i = 0; i < 4; ++i) {
    wmx_qkv[i]  = cvt(mx[i].qkv_w, 544, 1632);
    wmx_proj[i] = cvt(mx[i].proj_w, 544, 544);
    wmx_fc1[i]  = cvt(mx[i].fc1_w, 544, 1088);
    wmx_fc2[i]  = cvt(mx[i].fc2_w, 1088, 544);
    wmx_f1[i]   = cvt(mx[i].f1_w, 544, 1088);
    wmx_f2[i]   = cvt(mx[i].f2_w, 1088, 544);
  }
  k_init_dct<<<(27 * 81 + 255) / 256, 256, 0, stream>>>(dM2f, dM2, dM3);

  auto gemm = [&](const float* A, size_t wOff, const float* bias, const float* resid,
                  float* D, int M, int N, int K, int act) {
    if (N >= 128) {
      dim3 g((N + 127) / 128, (M + 127) / 128);
      k_gemm_wmma<128><<<g, 256, 0, stream>>>(A, W16 + wOff, bias, resid, D, M, N, K, act);
    } else {
      dim3 g((N + 63) / 64, (M + 127) / 128);
      k_gemm_wmma<64><<<g, 256, 0, stream>>>(A, W16 + wOff, bias, resid, D, M, N, K, act);
    }
  };
  auto ln = [&](const float* srcp, float* dstp, const float* g, const float* b,
                int rows, int C, float eps, int bd) {
    k_ln<<<rows, bd, 0, stream>>>(srcp, dstp, g, b, C, eps);
  };

  const int ROWS_SP = NBATCH * NK27 * NPTS;   // 235008
  const int ROWS_MX = NBATCH * SEQMX;         // 27648
  const int ROWS_HF = NBATCH * NK27;          // 13824
  const int ROWS_HD = NBATCH * NFRM;          // 41472
  const int EHALF = ROWS_HF * DIMT;           // 7520256

  // ---------------- spatial stage ----------------
  k_embed_sp<<<(ROWS_SP * DIMS + 255) / 256, 256, 0, stream>>>(x, joint_w, joint_b, spatial_pos, buf0);
  for (int i = 0; i < 4; ++i) {
    ln(buf0, buf1, sp[i].ln1_g, sp[i].ln1_b, ROWS_SP, DIMS, 1e-6f, 64);
    gemm(buf1, wsp_qkv[i], sp[i].qkv_b, nullptr, buf2, ROWS_SP, 96, 32, 0);
    k_attn_sp<<<NBATCH * NK27 * 8, 32, 0, stream>>>(buf2, buf1);
    gemm(buf1, wsp_proj[i], sp[i].proj_b, buf0, buf0, ROWS_SP, 32, 32, 0);
    ln(buf0, buf1, sp[i].ln2_g, sp[i].ln2_b, ROWS_SP, DIMS, 1e-6f, 64);
    gemm(buf1, wsp_fc1[i], sp[i].fc1_b, nullptr, buf3, ROWS_SP, 64, 32, 1);
    gemm(buf3, wsp_fc2[i], sp[i].fc2_b, buf0, buf0, ROWS_SP, 32, 64, 0);
  }
  ln(buf0, buf1, sp_norm_g, sp_norm_b, ROWS_SP, DIMS, 1e-6f, 64);  // buf1 = sf (B,27,544)

  // ---------------- freq embed + concat ----------------
  k_dct_frames<<<(ROWS_HF * 34 + 255) / 256, 256, 0, stream>>>(x, dM2f, buf2);
  gemm(buf2, w_freq, freq_b, nullptr, buf3, ROWS_HF, DIMT, 34, 0);
  k_concat_xt<<<(ROWS_MX * DIMT + 255) / 256, 256, 0, stream>>>(buf3, buf1, temp_pos_, temp_pos, buf4);

  // ---------------- mix blocks ----------------
  for (int i = 0; i < 4; ++i) {
    ln(buf4, buf2, mx[i].ln1_g, mx[i].ln1_b, ROWS_MX, DIMT, 1e-6f, 256);
    gemm(buf2, wmx_qkv[i], mx[i].qkv_b, nullptr, buf5, ROWS_MX, 1632, 544, 0);
    k_attn_mix<<<NBATCH * 8, 64, 0, stream>>>(buf5, buf2);
    gemm(buf2, wmx_proj[i], mx[i].proj_b, buf4, buf4, ROWS_MX, 544, 544, 0);
    // x1 = first half: standard MLP
    k_copy_half<<<(EHALF + 255) / 256, 256, 0, stream>>>(buf4, buf3, 0);
    ln(buf3, buf3b, mx[i].ln2_g, mx[i].ln2_b, ROWS_HF, DIMT, 1e-6f, 256);
    gemm(buf3b, wmx_fc1[i], mx[i].fc1_b, nullptr, buf6, ROWS_HF, 1088, 544, 1);
    gemm(buf6, wmx_fc2[i], mx[i].fc2_b, nullptr, buf7, ROWS_HF, 544, 1088, 0);
    k_add_half<<<(EHALF + 255) / 256, 256, 0, stream>>>(buf4, buf7, 0);
    // x2 = second half: LN -> DCT27 -> MLP -> IDCT27 -> add
    k_copy_half<<<(EHALF + 255) / 256, 256, 0, stream>>>(buf4, buf3, NK27);
    ln(buf3, buf3b, mx[i].ln3_g, mx[i].ln3_b, ROWS_HF, DIMT, 1e-6f, 256);
    k_seq_mat27<<<(EHALF + 255) / 256, 256, 0, stream>>>(buf3b, dM2, buf7);
    gemm(buf7, wmx_f1[i], mx[i].f1_b, nullptr, buf6, ROWS_HF, 1088, 544, 1);
    gemm(buf6, wmx_f2[i], mx[i].f2_b, nullptr, buf7b, ROWS_HF, 544, 1088, 0);
    k_seq_mat27<<<(EHALF + 255) / 256, 256, 0, stream>>>(buf7b, dM3, buf7);
    k_add_half<<<(EHALF + 255) / 256, 256, 0, stream>>>(buf4, buf7, NK27);
  }

  // ---------------- head ----------------
  ln(buf4, buf2, t_norm_g, t_norm_b, ROWS_MX, DIMT, 1e-6f, 256);
  k_patches<<<(ROWS_HD * 2 * DIMT + 255) / 256, 256, 0, stream>>>(buf2, patches_w, patches_b, buf5);
  ln(buf5, buf5, head_ln_g, head_ln_b, ROWS_HD, 2 * DIMT, 1e-5f, 256);
  gemm(buf5, w_head, head_b, nullptr, (float*)d_out, ROWS_HD, 51, 1088, 0);
}
